// SimpleAttention_48034914238768
// MI455X (gfx1250) — matrix operand order repaired and checked by reading, not compiled, not run on hardware
//
#include <hip/hip_runtime.h>
#include <hip/hip_bf16.h>

typedef __attribute__((ext_vector_type(16))) _Float16 v16h;
typedef __attribute__((ext_vector_type(8)))  _Float16 v8h;
typedef __attribute__((ext_vector_type(8)))  float    v8f;
typedef __attribute__((ext_vector_type(4)))  float    v4f;

constexpr int DIM = 4096;
constexpr int TC  = 32768;           // cached tokens (new token handled separately)
constexpr float SCALE = 0.015625f;   // 4096^-0.5

// ---- workspace layout (float offsets) ----
// q[4096] | k[4096] | v[4096] | s[32769] | pnew[1] | pad[2] | ph(f16,32768 = 16384 f32 slots) | partial[8*4096]
constexpr int WS_Q    = 0;
constexpr int WS_K    = 4096;
constexpr int WS_V    = 8192;
constexpr int WS_S    = 12288;
constexpr int WS_PNEW = 45057;
constexpr int WS_PH   = 45060;   // byte offset 180240, 16B aligned
constexpr int WS_PART = 61444;   // byte offset 245776, 16B aligned

__device__ __forceinline__ v16h pack16(v4f a, v4f b, v4f c, v4f d) {
  v16h r;
  r[0]=(_Float16)a[0];  r[1]=(_Float16)a[1];  r[2]=(_Float16)a[2];  r[3]=(_Float16)a[3];
  r[4]=(_Float16)b[0];  r[5]=(_Float16)b[1];  r[6]=(_Float16)b[2];  r[7]=(_Float16)b[3];
  r[8]=(_Float16)c[0];  r[9]=(_Float16)c[1];  r[10]=(_Float16)c[2]; r[11]=(_Float16)c[3];
  r[12]=(_Float16)d[0]; r[13]=(_Float16)d[1]; r[14]=(_Float16)d[2]; r[15]=(_Float16)d[3];
  return r;
}

// Dot of one shared vector `vec` (f32, len DIM) against 16 consecutive rows of
// row-major f32 matrix `mat`, via V_WMMA_F32_16X16X32_F16.
// A = vec broadcast across the 16 M-rows; B columns = matrix rows row0..row0+15.
// Returns: on lanes 0..15, the dot for row (row0 + lane).
__device__ __forceinline__ float dot_rows16_wmma(const float* __restrict__ vec,
                                                 const float* __restrict__ mat,
                                                 int row0, int lane) {
  const int hh = (lane >> 4) & 1;   // wave half
  const int ln = lane & 15;
  const float* rowp = mat + (size_t)(row0 + ln) * DIM + hh * 8;  // B: K pattern per half
  const float* vp   = vec + hh * 8;                              // A: K pattern per half
  v8f acc = {};
  for (int c = 0; c < DIM / 32; ++c) {
    const int base = c * 32;
    // A fragment: elements 0..7 = vec[base+off+0..7], 8..15 = vec[base+off+16..23]
    v4f a0 = *(const v4f*)(vp + base + 0);
    v4f a1 = *(const v4f*)(vp + base + 4);
    v4f a2 = *(const v4f*)(vp + base + 16);
    v4f a3 = *(const v4f*)(vp + base + 20);
    // B fragment: elements 0..7 = mat[row0+ln][base+off+0..7], 8..15 = mat[row0+ln][base+off+16..23]
    v4f b0 = *(const v4f*)(rowp + base + 0);
    v4f b1 = *(const v4f*)(rowp + base + 4);
    v4f b2 = *(const v4f*)(rowp + base + 16);
    v4f b3 = *(const v4f*)(rowp + base + 20);
    v16h A = pack16(a0, a1, a2, a3);
    v16h B = pack16(b0, b1, b2, b3);
    acc = __builtin_amdgcn_wmma_f32_16x16x32_f16(false, A, false, B, (short)0, acc, false, false);
  }
  return acc[0];  // D[M=0][N=lane] on lanes 0..15
}

// K1: q/k/v projections. 12288 rows, 16 rows per wave -> 768 waves -> 96 blocks.
__global__ void __launch_bounds__(256) qkv_gemv_kernel(
    const float* __restrict__ x,
    const float* __restrict__ Wq, const float* __restrict__ bq,
    const float* __restrict__ Wk, const float* __restrict__ bk,
    const float* __restrict__ Wv, const float* __restrict__ bv,
    float* __restrict__ qo, float* __restrict__ ko, float* __restrict__ vo) {
  const int wave = blockIdx.x * 8 + (threadIdx.x >> 5);
  const int lane = threadIdx.x & 31;
  const int row0 = wave * 16;
  const float* W; const float* b; float* o; int r;
  if (row0 < DIM)            { W = Wq; b = bq; o = qo; r = row0; }
  else if (row0 < 2 * DIM)   { W = Wk; b = bk; o = ko; r = row0 - DIM; }
  else                       { W = Wv; b = bv; o = vo; r = row0 - 2 * DIM; }
  float d = dot_rows16_wmma(x, W, r, lane);
  if (lane < 16) o[r + lane] = d + b[r + lane];
}

// K2: logits over K_cache. 32768 rows -> 2048 waves -> 256 blocks.
__global__ void __launch_bounds__(256) logits_kernel(
    const float* __restrict__ q, const float* __restrict__ Kc,
    float* __restrict__ s) {
  const int wave = blockIdx.x * 8 + (threadIdx.x >> 5);
  const int lane = threadIdx.x & 31;
  const int r0 = wave * 16;
  float d = dot_rows16_wmma(q, Kc, r0, lane);
  if (lane < 16) s[r0 + lane] = d * SCALE;
}

// K3: s[32768] = q.k*scale, then softmax over 32769; write f16 probs (x1024) + pnew.
__global__ void __launch_bounds__(1024) softmax_kernel(
    float* __restrict__ s, const float* __restrict__ q, const float* __restrict__ k,
    _Float16* __restrict__ ph, float* __restrict__ pnew) {
  __shared__ float red[1024];
  const int tid = threadIdx.x;
  // 1) new-token logit
  float part = 0.f;
  #pragma unroll
  for (int j = 0; j < 4; ++j) { int i = tid + j * 1024; part += q[i] * k[i]; }
  red[tid] = part; __syncthreads();
  for (int o = 512; o > 0; o >>= 1) { if (tid < o) red[tid] += red[tid + o]; __syncthreads(); }
  if (tid == 0) s[TC] = red[0] * SCALE;
  __syncthreads();
  // 2) max
  float m = -3.402823466e38f;
  for (int t = tid; t < TC + 1; t += 1024) m = fmaxf(m, s[t]);
  red[tid] = m; __syncthreads();
  for (int o = 512; o > 0; o >>= 1) { if (tid < o) red[tid] = fmaxf(red[tid], red[tid + o]); __syncthreads(); }
  const float M = red[0]; __syncthreads();
  // 3) exp + sum (store exp back into s)
  float sum = 0.f;
  for (int t = tid; t < TC + 1; t += 1024) { float e = __expf(s[t] - M); s[t] = e; sum += e; }
  red[tid] = sum; __syncthreads();
  for (int o = 512; o > 0; o >>= 1) { if (tid < o) red[tid] += red[tid + o]; __syncthreads(); }
  const float Z = red[0]; __syncthreads();
  // 4) normalize; scale by 1024 so f16 probs stay in the normal range
  const float a = 1024.0f / Z;
  for (int t = tid; t < TC; t += 1024) ph[t] = (_Float16)(s[t] * a);
  if (tid == 0) pnew[0] = s[TC] / Z;
}

// K4: out_partial = p @ V_cache. Wave = (d-chunk, t-slice); 256x8 waves -> 256 blocks.
// A = probs broadcast; B column n = V[t..t+31][d0+n] (64B coalesced per half-wave).
__global__ void __launch_bounds__(256) attn_v_kernel(
    const float* __restrict__ Vc, const _Float16* __restrict__ ph,
    float* __restrict__ partial) {
  const int wave  = blockIdx.x * 8 + (threadIdx.x >> 5);
  const int lane  = threadIdx.x & 31;
  const int d0    = (wave >> 3) * 16;
  const int slice = wave & 7;
  const int hh    = (lane >> 4) & 1;
  const int ln    = lane & 15;
  v8f acc = {};
  for (int c = slice; c < TC / 32; c += 8) {
    const int tb = c * 32;
    const _Float16* pp = ph + tb + hh * 8;
    v8h p0 = *(const v8h*)(pp);
    v8h p1 = *(const v8h*)(pp + 16);
    v16h A;
    #pragma unroll
    for (int i = 0; i < 8; ++i) { A[i] = p0[i]; A[8 + i] = p1[i]; }
    const float* vb = Vc + (size_t)(tb + hh * 8) * DIM + d0 + ln;
    v16h B;
    #pragma unroll
    for (int kk = 0; kk < 16; ++kk) B[kk] = (_Float16)vb[(size_t)(kk < 8 ? kk : kk + 8) * DIM];
    acc = __builtin_amdgcn_wmma_f32_16x16x32_f16(false, A, false, B, (short)0, acc, false, false);
  }
  if (lane < 16) partial[slice * DIM + d0 + lane] = acc[0];
}

// K5: reduce 8 partials, un-scale the x1024, add the new-token contribution p_new * v.
__global__ void __launch_bounds__(256) finalize_kernel(
    const float* __restrict__ partial, const float* __restrict__ pnew,
    const float* __restrict__ v, float* __restrict__ out) {
  const int i = blockIdx.x * blockDim.x + threadIdx.x;  // 4096 total
  float s = 0.f;
  #pragma unroll
  for (int p = 0; p < 8; ++p) s += partial[p * DIM + i];
  out[i] = s * (1.0f / 1024.0f) + pnew[0] * v[i];
}

extern "C" void kernel_launch(void* const* d_in, const int* in_sizes, int n_in,
                              void* d_out, int out_size, void* d_ws, size_t ws_size,
                              hipStream_t stream) {
  const float* x  = (const float*)d_in[0];
  const float* Kc = (const float*)d_in[1];
  const float* Vc = (const float*)d_in[2];
  const float* Wq = (const float*)d_in[3];
  const float* bq = (const float*)d_in[4];
  const float* Wk = (const float*)d_in[5];
  const float* bk = (const float*)d_in[6];
  const float* Wv = (const float*)d_in[7];
  const float* bv = (const float*)d_in[8];
  float* out = (float*)d_out;

  float* w = (float*)d_ws;
  float*     q       = w + WS_Q;
  float*     k       = w + WS_K;
  float*     v       = w + WS_V;
  float*     s       = w + WS_S;
  float*     pnew    = w + WS_PNEW;
  _Float16*  ph      = (_Float16*)(w + WS_PH);
  float*     partial = w + WS_PART;

  qkv_gemv_kernel<<<96, 256, 0, stream>>>(x, Wq, bq, Wk, bk, Wv, bv, q, k, v);
  logits_kernel<<<256, 256, 0, stream>>>(q, Kc, s);
  softmax_kernel<<<1, 1024, 0, stream>>>(s, q, k, ph, pnew);
  attn_v_kernel<<<256, 256, 0, stream>>>(Vc, ph, partial);
  finalize_kernel<<<16, 256, 0, stream>>>(partial, pnew, v, out);
}
